// ModularEncoder_71502615544595
// MI455X (gfx1250) — matrix operand order repaired and checked by reading, not compiled, not run on hardware
//
#include <hip/hip_runtime.h>
#include <hip/hip_bf16.h>

// Sizes from the reference
#define NB   8192   // batch
#define ND   1024   // input dim
#define NH   400    // hidden
#define NK2  512    // 2*L
#define NE   16     // experts

#define BSTRIDE 40  // padded N-major LDS stride for B tiles (80B rows: 16B-aligned, bank-spread)

typedef __attribute__((ext_vector_type(16))) __bf16 v16bf;
typedef __attribute__((ext_vector_type(8)))  __bf16 bf16x8;
typedef __attribute__((ext_vector_type(4)))  __bf16 bf16x4;
typedef __attribute__((ext_vector_type(8)))  float  v8f;

#define WMMA_BF16(a, b, c) \
  __builtin_amdgcn_wmma_f32_16x16x32_bf16(false, (a), false, (b), (short)0, (c), false, false)

// ---- fp32 -> bf16 hi/lo split (hi+lo carries ~16 mantissa bits) ----
__device__ inline __bf16 bf_hi(float v) { return (__bf16)v; }
__device__ inline __bf16 bf_lo(float v, __bf16 h) { return (__bf16)(v - (float)h); }

__device__ inline void split4(const float4& v, bf16x4& hi, bf16x4& lo) {
  __bf16 h0 = bf_hi(v.x), h1 = bf_hi(v.y), h2 = bf_hi(v.z), h3 = bf_hi(v.w);
  hi[0] = h0; hi[1] = h1; hi[2] = h2; hi[3] = h3;
  lo[0] = bf_lo(v.x, h0); lo[1] = bf_lo(v.y, h1);
  lo[2] = bf_lo(v.z, h2); lo[3] = bf_lo(v.w, h3);
}

__device__ inline unsigned pack2(__bf16 h, __bf16 l) {
  unsigned short hs = __builtin_bit_cast(unsigned short, h);
  unsigned short ls = __builtin_bit_cast(unsigned short, l);
  return ((unsigned)ls << 16) | (unsigned)hs;
}

__device__ inline __bf16 lo16(unsigned u) {
  return __builtin_bit_cast(__bf16, (unsigned short)(u & 0xffffu));
}
__device__ inline __bf16 hi16(unsigned u) {
  return __builtin_bit_cast(__bf16, (unsigned short)(u >> 16));
}

// A fragment: 16x32 bf16 tile from LDS (row-major, stride 32 elements).
// ISA: lanes 0-15 row M=lane, lanes 16-31 row M=lane-16;
// element j -> K = j + 8*(j>=8) + 8*(lane>=16): two contiguous 8-elem runs.
__device__ inline v16bf loadA(const __bf16* lds, int mi, int lane) {
  int row  = mi * 16 + (lane & 15);
  int kadd = (lane >> 4) * 8;
  const __bf16* base = lds + row * 32 + kadd;
  bf16x8 a0 = *(const bf16x8*)(base);        // K = kadd .. kadd+7
  bf16x8 a1 = *(const bf16x8*)(base + 16);   // K = kadd+16 .. kadd+23
  return __builtin_shufflevector(a0, a1, 0,1,2,3,4,5,6,7,8,9,10,11,12,13,14,15);
}

// B fragment from N-major (transposed) LDS tile, stride BSTRIDE.
// element j -> K = j + 8*(j>=8) + 8*(lane>=16); lane%16 = N column.
__device__ inline v16bf loadB(const __bf16* lds, int nj, int lane) {
  int col  = nj * 16 + (lane & 15);
  int kadd = (lane >> 4) * 8;
  const __bf16* base = lds + col * BSTRIDE + kadd;
  bf16x8 b0 = *(const bf16x8*)(base);        // K = kadd .. kadd+7
  bf16x8 b1 = *(const bf16x8*)(base + 16);   // K = kadd+16 .. kadd+23
  return __builtin_shufflevector(b0, b1, 0,1,2,3,4,5,6,7,8,9,10,11,12,13,14,15);
}

// ---------------- grouping: histogram / scan / scatter ----------------
__global__ void zero_counts_kernel(int* ws_i) {
  if (threadIdx.x < 64) ws_i[threadIdx.x] = 0;
}

__global__ void hist_kernel(const int* __restrict__ idx, int* __restrict__ counts) {
  int b = blockIdx.x * blockDim.x + threadIdx.x;
  if (b < NB) atomicAdd(&counts[idx[b] & (NE - 1)], 1);
}

__global__ void scan_kernel(int* ws_i) {
  if (threadIdx.x == 0) {
    int acc = 0;
    for (int e = 0; e < NE; ++e) {
      ws_i[16 + e] = acc;   // offsets[e]
      ws_i[33 + e] = acc;   // cursor[e]
      acc += ws_i[e];
    }
    ws_i[32] = acc;         // offsets[16] == NB
  }
}

__global__ void scatter_kernel(const int* __restrict__ idx, int* __restrict__ cursor,
                               int* __restrict__ perm) {
  int b = blockIdx.x * blockDim.x + threadIdx.x;
  if (b < NB) {
    int e = idx[b] & (NE - 1);
    int pos = atomicAdd(&cursor[e], 1);
    perm[pos] = b;
  }
}

// ---------------- GEMM1: h1 = relu(x[perm] @ W1[e] + b1[e]) ----------------
// grid: (128 row-tiles, 16 experts, 7 col-tiles of 64), block = 256 (8 waves)
__global__ __launch_bounds__(256) void gemm1_kernel(
    const float* __restrict__ x, const float* __restrict__ W1, const float* __restrict__ b1,
    const int* __restrict__ offsets, const int* __restrict__ perm,
    unsigned* __restrict__ h1p) {
  const int e   = blockIdx.y;
  const int off = offsets[e];
  const int cnt = offsets[e + 1] - off;
  const int m0  = blockIdx.x * 64;
  if (m0 >= cnt) return;                       // uniform per block
  const int n0  = blockIdx.z * 64;
  int mcount = cnt - m0; if (mcount > 64) mcount = 64;

  __shared__ __bf16 Ah[64 * 32], Al[64 * 32];
  __shared__ __bf16 Bh[64 * BSTRIDE], Bl[64 * BSTRIDE];   // N-major
  __shared__ int rows_s[64];

  const int tid  = threadIdx.x;
  const int lane = tid & 31;
  const int w    = tid >> 5;
  const int nj   = w & 3;      // N sub-tile (16 cols)
  const int mi0  = w >> 2;     // M sub-tiles mi0 and mi0+2

  if (tid < 64) rows_s[tid] = (tid < mcount) ? perm[off + m0 + tid] : -1;
  __syncthreads();

  v8f acc0 = {}; v8f acc1 = {};
  const float* W1e = W1 + (long)e * ND * NH;

  for (int k0 = 0; k0 < ND; k0 += 32) {
    // stage A (64x32 floats) gathered rows of x: float4 per slot, bf16x4 stores
#pragma unroll
    for (int i = 0; i < 2; ++i) {
      int s = tid + i * 256;           // 512 float4 slots
      int row = s >> 3, kq = (s & 7) * 4;
      float4 v = make_float4(0.f, 0.f, 0.f, 0.f);
      int r = rows_s[row];
      if (r >= 0) {
        const float* p = x + (long)r * ND + k0 + kq;
        v = *(const float4*)p;
        if (k0 + 32 < ND) __builtin_prefetch(p + 32, 0, 3);
      }
      bf16x4 hi, lo; split4(v, hi, lo);
      *(bf16x4*)(Ah + row * 32 + kq) = hi;
      *(bf16x4*)(Al + row * 32 + kq) = lo;
    }
    // stage B (32x64 floats) slice of W1[e]: float4 loads, transposed b16 stores
#pragma unroll
    for (int i = 0; i < 2; ++i) {
      int s = tid + i * 256;           // 512 float4 slots
      int kk = s >> 4, col4 = (s & 15) * 4;
      float4 v = make_float4(0.f, 0.f, 0.f, 0.f);
      if (n0 + col4 < NH) {
        const float* p = W1e + (long)(k0 + kk) * NH + n0 + col4;
        v = *(const float4*)p;
        if (k0 + 32 < ND) __builtin_prefetch(p + 32 * NH, 0, 3);
      }
      bf16x4 hi, lo; split4(v, hi, lo);
#pragma unroll
      for (int c = 0; c < 4; ++c) {
        Bh[(col4 + c) * BSTRIDE + kk] = hi[c];
        Bl[(col4 + c) * BSTRIDE + kk] = lo[c];
      }
    }
    __syncthreads();

    v16bf bh = loadB(Bh, nj, lane);
    v16bf bl = loadB(Bl, nj, lane);
    {
      v16bf ah = loadA(Ah, mi0, lane);
      v16bf al = loadA(Al, mi0, lane);
      acc0 = WMMA_BF16(ah, bh, acc0);
      acc0 = WMMA_BF16(ah, bl, acc0);
      acc0 = WMMA_BF16(al, bh, acc0);
    }
    {
      v16bf ah = loadA(Ah, mi0 + 2, lane);
      v16bf al = loadA(Al, mi0 + 2, lane);
      acc1 = WMMA_BF16(ah, bh, acc1);
      acc1 = WMMA_BF16(ah, bl, acc1);
      acc1 = WMMA_BF16(al, bh, acc1);
    }
    __syncthreads();
  }

  // epilogue: +bias, relu, store h1 (grouped order) as packed (hi|lo) uint
  int col = n0 + nj * 16 + (lane & 15);
  if (col < NH) {
    float bias = b1[e * NH + col];
#pragma unroll
    for (int r = 0; r < 8; ++r) {
      int rlo = r + ((lane >> 4) << 3);
      int row0 = mi0 * 16 + rlo;
      int row1 = (mi0 + 2) * 16 + rlo;
      if (row0 < mcount) {
        float v = acc0[r] + bias; v = v > 0.f ? v : 0.f;
        __bf16 h = bf_hi(v);
        h1p[(long)(off + m0 + row0) * NH + col] = pack2(h, bf_lo(v, h));
      }
      if (row1 < mcount) {
        float v = acc1[r] + bias; v = v > 0.f ? v : 0.f;
        __bf16 h = bf_hi(v);
        h1p[(long)(off + m0 + row1) * NH + col] = pack2(h, bf_lo(v, h));
      }
    }
  }
}

// ---------------- GEMM2: out = h1 @ W2[e] + b2[e], scatter mu/logvar ----------------
// grid: (128 row-tiles, 16 experts, 4 col-tiles of 128), block = 256 (8 waves)
__global__ __launch_bounds__(256) void gemm2_kernel(
    const unsigned* __restrict__ h1p,
    const float* __restrict__ W2, const float* __restrict__ b2,
    const int* __restrict__ offsets, const int* __restrict__ perm,
    float* __restrict__ out) {
  const int e   = blockIdx.y;
  const int off = offsets[e];
  const int cnt = offsets[e + 1] - off;
  const int m0  = blockIdx.x * 64;
  if (m0 >= cnt) return;
  const int n0  = blockIdx.z * 128;
  int mcount = cnt - m0; if (mcount > 64) mcount = 64;

  __shared__ __bf16 Ah[64 * 32],       Al[64 * 32];
  __shared__ __bf16 Bh[128 * BSTRIDE], Bl[128 * BSTRIDE];   // N-major
  __shared__ int rows_s[64];

  const int tid  = threadIdx.x;
  const int lane = tid & 31;
  const int w    = tid >> 5;
  const int mi   = w & 3;       // M sub-tile
  const int ng   = w >> 2;      // 64-col group within the 128-col tile

  if (tid < 64) rows_s[tid] = (tid < mcount) ? perm[off + m0 + tid] : -1;

  v8f acc[4] = {};
  const float* W2e = W2 + (long)e * NH * NK2;

  for (int k0 = 0; k0 < NH; k0 += 32) {        // 13 steps; last zero-padded
    // stage A (64x32 packed uints) from h1p: uint4 loads, b64 hi/lo stores
#pragma unroll
    for (int i = 0; i < 2; ++i) {
      int s = tid + i * 256;           // 512 uint4 slots
      int row = s >> 3, kq = (s & 7) * 4;
      uint4 v = make_uint4(0u, 0u, 0u, 0u);
      if (row < mcount && (k0 + kq) < NH)
        v = *(const uint4*)(h1p + (long)(off + m0 + row) * NH + k0 + kq);
      bf16x4 hi, lo;
      hi[0] = lo16(v.x); lo[0] = hi16(v.x);
      hi[1] = lo16(v.y); lo[1] = hi16(v.y);
      hi[2] = lo16(v.z); lo[2] = hi16(v.z);
      hi[3] = lo16(v.w); lo[3] = hi16(v.w);
      *(bf16x4*)(Ah + row * 32 + kq) = hi;
      *(bf16x4*)(Al + row * 32 + kq) = lo;
    }
    // stage B (32x128 floats) slice of W2[e]: float4 loads, transposed stores
#pragma unroll
    for (int i = 0; i < 4; ++i) {
      int s = tid + i * 256;           // 1024 float4 slots
      int kk = s >> 5, col4 = (s & 31) * 4;
      float4 v = make_float4(0.f, 0.f, 0.f, 0.f);
      if ((k0 + kk) < NH) {
        const float* p = W2e + (long)(k0 + kk) * NK2 + n0 + col4;
        v = *(const float4*)p;
        if (k0 + 32 < NH) __builtin_prefetch(p + 32 * NK2, 0, 3);
      }
      bf16x4 hi, lo; split4(v, hi, lo);
#pragma unroll
      for (int c = 0; c < 4; ++c) {
        Bh[(col4 + c) * BSTRIDE + kk] = hi[c];
        Bl[(col4 + c) * BSTRIDE + kk] = lo[c];
      }
    }
    __syncthreads();

    v16bf ah = loadA(Ah, mi, lane);
    v16bf al = loadA(Al, mi, lane);
#pragma unroll
    for (int t = 0; t < 4; ++t) {
      int nj = ng * 4 + t;
      v16bf bh = loadB(Bh, nj, lane);
      v16bf bl = loadB(Bl, nj, lane);
      acc[t] = WMMA_BF16(ah, bh, acc[t]);
      acc[t] = WMMA_BF16(ah, bl, acc[t]);
      acc[t] = WMMA_BF16(al, bh, acc[t]);
    }
    __syncthreads();
  }

  // epilogue: +b2, scatter to z_mu / z_logvar halves of out
#pragma unroll
  for (int t = 0; t < 4; ++t) {
    int col = n0 + (ng * 4 + t) * 16 + (lane & 15);
    float bias = b2[e * NK2 + col];
#pragma unroll
    for (int r = 0; r < 8; ++r) {
      int row = mi * 16 + r + ((lane >> 4) << 3);
      if (row < mcount) {
        int b = rows_s[row];
        float v = acc[t][r] + bias;
        if (col < 256) out[(long)b * 256 + col] = v;
        else           out[(long)NB * 256 + (long)b * 256 + (col - 256)] = v;
      }
    }
  }
}

extern "C" void kernel_launch(void* const* d_in, const int* in_sizes, int n_in,
                              void* d_out, int out_size, void* d_ws, size_t ws_size,
                              hipStream_t stream) {
  (void)in_sizes; (void)n_in; (void)out_size; (void)ws_size;
  const float* x     = (const float*)d_in[0];
  const int*   angle = (const int*)d_in[1];
  const float* W1    = (const float*)d_in[2];
  const float* b1    = (const float*)d_in[3];
  const float* W2    = (const float*)d_in[4];
  const float* b2    = (const float*)d_in[5];
  float* out = (float*)d_out;

  // ws layout: [0..15] counts, [16..32] offsets, [33..48] cursor, pad to 64 ints,
  // then perm[8192] ints, then packed h1 (hi|lo) uints (8192*400).
  int* ws_i = (int*)d_ws;
  int* perm = ws_i + 64;
  unsigned* h1p = (unsigned*)((char*)d_ws + 64 * 4 + NB * 4);

  zero_counts_kernel<<<1, 64, 0, stream>>>(ws_i);
  hist_kernel<<<NB / 256, 256, 0, stream>>>(angle, ws_i);
  scan_kernel<<<1, 32, 0, stream>>>(ws_i);
  scatter_kernel<<<NB / 256, 256, 0, stream>>>(angle, ws_i + 33, perm);

  dim3 g1(NB / 64, NE, (NH + 63) / 64);   // 128 x 16 x 7, most blocks early-exit
  gemm1_kernel<<<g1, 256, 0, stream>>>(x, W1, b1, ws_i + 16, perm, h1p);

  dim3 g2(NB / 64, NE, NK2 / 128);        // 128 x 16 x 4
  gemm2_kernel<<<g2, 256, 0, stream>>>(h1p, W2, b2, ws_i + 16, perm, out);
}
